// gMLPBLOCK_CLS_55155970015902
// MI455X (gfx1250) — compile-verified, matrix operand order checked by reading, not run on hardware
//
#include <hip/hip_runtime.h>
#include <hip/hip_bf16.h>
#include <math.h>

typedef unsigned int u32;
typedef unsigned short u16;
typedef __attribute__((ext_vector_type(16))) __bf16 v16bf;
typedef __attribute__((ext_vector_type(8)))  float  v8f;
typedef __attribute__((ext_vector_type(8)))  unsigned int v8u;

// ---------- helpers ----------
__device__ __forceinline__ u16 f2bf(float f) {
    u32 u = __builtin_bit_cast(u32, f);
    u32 r = u + 0x7FFFu + ((u >> 16) & 1u);   // round-to-nearest-even
    return (u16)(r >> 16);
}
__device__ __forceinline__ float bf2f(u16 h) {
    u32 u = ((u32)h) << 16;
    return __builtin_bit_cast(float, u);
}
__device__ __forceinline__ float gelu_erf(float x) {
    return 0.5f * x * (1.0f + erff(x * 0.70710678118654752f));
}
__device__ __forceinline__ float block_sum(float v, float* sdata) {
    const int tid = threadIdx.x;
    sdata[tid] = v;
    __syncthreads();
    for (int s = 128; s > 0; s >>= 1) {
        if (tid < s) sdata[tid] += sdata[tid + s];
        __syncthreads();
    }
    float r = sdata[0];
    __syncthreads();
    return r;
}

// ---------- elementwise fp32 -> bf16 convert ----------
__global__ void cvt_f32_bf16(const float* __restrict__ in, u16* __restrict__ out, int n) {
    int i = blockIdx.x * 256 + threadIdx.x;
    if (i < n) out[i] = f2bf(in[i]);
}

// ---------- LayerNorm, fp32 input -> bf16 output ----------
__global__ void ln_f32(const float* __restrict__ x, const float* __restrict__ g,
                       const float* __restrict__ b, u16* __restrict__ out, int D, int per) {
    __shared__ float sdata[256];
    const int row = blockIdx.x, tid = threadIdx.x;
    const float* xr = x + (size_t)row * D;
    float vals[12];
    float s = 0.f;
    for (int i = 0; i < per; ++i) { vals[i] = xr[tid + i * 256]; s += vals[i]; }
    float mean = block_sum(s, sdata) / (float)D;
    float q = 0.f;
    for (int i = 0; i < per; ++i) { float d = vals[i] - mean; q += d * d; }
    float var = block_sum(q, sdata) / (float)D;
    float inv = rsqrtf(var + 1e-5f);
    u16* orow = out + (size_t)row * D;
    for (int i = 0; i < per; ++i) {
        int j = tid + i * 256;
        orow[j] = f2bf((vals[i] - mean) * inv * g[j] + b[j]);
    }
}

// ---------- LayerNorm, bf16 input -> bf16 output ----------
__global__ void ln_bf16(const u16* __restrict__ x, const float* __restrict__ g,
                        const float* __restrict__ b, u16* __restrict__ out, int D, int per) {
    __shared__ float sdata[256];
    const int row = blockIdx.x, tid = threadIdx.x;
    const u16* xr = x + (size_t)row * D;
    float vals[12];
    float s = 0.f;
    for (int i = 0; i < per; ++i) { vals[i] = bf2f(xr[tid + i * 256]); s += vals[i]; }
    float mean = block_sum(s, sdata) / (float)D;
    float q = 0.f;
    for (int i = 0; i < per; ++i) { float d = vals[i] - mean; q += d * d; }
    float var = block_sum(q, sdata) / (float)D;
    float inv = rsqrtf(var + 1e-5f);
    u16* orow = out + (size_t)row * D;
    for (int i = 0; i < per; ++i) {
        int j = tid + i * 256;
        orow[j] = f2bf((vals[i] - mean) * inv * g[j] + b[j]);
    }
}

// ---------- cls routing: route[b] = (tanh(hn[b,0,:]@Wc + bc) < 0) ----------
__global__ void cls_route(const u16* __restrict__ hn, const float* __restrict__ Wc,
                          const float* __restrict__ bc, int* __restrict__ route,
                          int S, int F) {
    __shared__ float sdata[256];
    const int bz = blockIdx.x, tid = threadIdx.x;
    const u16* r0 = hn + (size_t)bz * S * F;   // sequence position 0
    float s = 0.f;
    for (int f = tid; f < F; f += 256) s += bf2f(r0[f]) * Wc[f];
    float tot = block_sum(s, sdata);
    if (tid == 0) {
        float c = tanhf(tot + bc[0]);
        route[bz] = (c < 0.f) ? 1 : 0;
    }
}

// ---------- LDS fragment load matching CDNA5 16-bit A/B WMMA layout ----------
// stride = 72 u16 per row (144B, 16B aligned); within a 32-wide k-step:
// lanes 0-15 take K 0..7 & 16..23, lanes 16-31 take K 8..15 & 24..31
__device__ __forceinline__ v16bf load_frag(const u16* base, int rc, int lane, int ks) {
    const u16* p = base + rc * 72 + ks * 32 + ((lane >> 4) << 3);
    uint4 lo = *(const uint4*)(p);        // 8 bf16
    uint4 hi = *(const uint4*)(p + 16);   // 8 bf16
    v8u u;
    u[0] = lo.x; u[1] = lo.y; u[2] = lo.z; u[3] = lo.w;
    u[4] = hi.x; u[5] = hi.y; u[6] = hi.z; u[7] = hi.w;
    return __builtin_bit_cast(v16bf, u);
}

// ---------- split staging: A-phase / B-phase, 16 VGPRs live each ----------
__device__ __forceinline__ void loadA(const u16* __restrict__ Ap, int K, int tm0, int k0,
                                      int tid, uint4 ra[4]) {
    #pragma unroll
    for (int it = 0; it < 4; ++it) {       // A: 128 x 64 bf16, 8 vec/row
        int v = tid + it * 256;
        u32 off = (u32)(tm0 + (v >> 3)) * (u32)K + (u32)(k0 + ((v & 7) << 3));
        ra[it] = *(const uint4*)(Ap + off);
    }
}
__device__ __forceinline__ void loadB(const u16* __restrict__ Bp, int N, int tn0, int k0,
                                      int tid, uint4 rb[4]) {
    #pragma unroll
    for (int it = 0; it < 4; ++it) {       // B: 64 x 128 bf16, 16 vec/row
        int v = tid + it * 256;
        u32 off = (u32)(k0 + (v >> 4)) * (u32)N + (u32)(tn0 + ((v & 15) << 3));
        rb[it] = *(const uint4*)(Bp + off);
    }
}
__device__ __forceinline__ void storeA(u16* __restrict__ la, int tid, const uint4 ra[4]) {
    #pragma unroll
    for (int it = 0; it < 4; ++it) {
        int v = tid + it * 256;
        *(uint4*)(&la[(v >> 3) * 72 + ((v & 7) << 3)]) = ra[it];
    }
}
__device__ __forceinline__ void storeB(u16* __restrict__ lb, int tid, const uint4 rb[4]) {
    #pragma unroll
    for (int it = 0; it < 4; ++it) {
        int v = tid + it * 256;
        int k = v >> 4, n = (v & 15) << 3;
        uint4 q = rb[it];
        u16 e[8] = { (u16)(q.x), (u16)(q.x >> 16), (u16)(q.y), (u16)(q.y >> 16),
                     (u16)(q.z), (u16)(q.z >> 16), (u16)(q.w), (u16)(q.w >> 16) };
        #pragma unroll
        for (int j = 0; j < 8; ++j) lb[(n + j) * 72 + k] = e[j];
    }
}

// one 32-wide k-step: 4x2 tiles = 8 WMMAs
__device__ __forceinline__ void compute_ks(const u16* la, const u16* lb,
                                           int wm, int wn, int lane, int ks, v8f acc[4][2]) {
    v16bf af[4], bfr[2];
    #pragma unroll
    for (int mf = 0; mf < 4; ++mf)
        af[mf] = load_frag(la, wm * 64 + mf * 16 + (lane & 15), lane, ks);
    #pragma unroll
    for (int nf = 0; nf < 2; ++nf)
        bfr[nf] = load_frag(lb, wn * 32 + nf * 16 + (lane & 15), lane, ks);
    #pragma unroll
    for (int mf = 0; mf < 4; ++mf)
        #pragma unroll
        for (int nf = 0; nf < 2; ++nf)
            acc[mf][nf] = __builtin_amdgcn_wmma_f32_16x16x32_bf16(
                false, af[mf], false, bfr[nf], (short)0, acc[mf][nf], false, false);
}

// ---------- generic bf16 WMMA GEMM, 128x128x64 block tile, double-buffered LDS ----------
// mode 0: out_bf[r,c]   = bf16( gelu(acc + bias[c]) )                         (GEMM1)
// mode 1: out_bf[b,r,c] = bf16( gelu(acc + rowbias[r] + sgu_res[b,r,c]) )     (spatial, A selected by route[b])
// mode 2: out_f32[r,c]  = res_f32[r,c] + gelu(acc + bias[c])                  (GEMM2)
__global__ __launch_bounds__(256, 1)
void gemm_bf16_wmma(const u16* __restrict__ A, const u16* __restrict__ Bmat,
                    int M, int N, int K, int mode,
                    const float* __restrict__ bias,
                    const u16* __restrict__ res_bf,
                    const float* __restrict__ res_f32,
                    u16* __restrict__ out_bf,
                    float* __restrict__ out_f32,
                    const u16* __restrict__ spA_i, const u16* __restrict__ spA_ii,
                    const float* __restrict__ bp_i, const float* __restrict__ bp_ii,
                    const int* __restrict__ route) {
    __shared__ u16 ldsA[2][128 * 72];   // 2 x 18KB
    __shared__ u16 ldsB[2][128 * 72];   // 2 x 18KB

    const int tid  = threadIdx.x;
    const int lane = tid & 31;
    const int wid  = tid >> 5;
    const int wm   = wid & 1;     // 0..1 -> 64-row slab
    const int wn   = wid >> 1;    // 0..3 -> 32-col slab
    const int tn0  = blockIdx.x * 128;
    const int tm0  = blockIdx.y * 128;
    const int bz   = blockIdx.z;

    const u16* Ap = A;
    const u16* Bp = Bmat;
    const float* rowbias = nullptr;
    if (mode == 1) {
        const int rt = route[bz];
        Ap      = rt ? spA_i : spA_ii;
        rowbias = rt ? bp_i  : bp_ii;
        Bp      = Bmat + (size_t)bz * (size_t)K * (size_t)N;
    }

    v8f acc[4][2] = {};
    const int nk = K >> 6;    // K / 64; all K multiples of 64, nk even
    uint4 ra[4], rb[4];

    // prologue: tile 0 -> LDS[0]
    loadA(Ap, K, tm0, 0, tid, ra);
    loadB(Bp, N, tn0, 0, tid, rb);
    storeA(&ldsA[0][0], tid, ra);
    storeB(&ldsB[0][0], tid, rb);
    __syncthreads();

    // unrolled-by-2 main loop; staging is UNconditional (last prefetch clamps to k0=0,
    // a dummy copy into the buffer nobody reads after the final barrier)
    for (int kt = 0; kt < nk; kt += 2) {
        const int k1 = (kt + 1) << 6;
        const int k2 = ((kt + 2) < nk) ? ((kt + 2) << 6) : 0;

        // half 1: compute buffer 0, stage tile kt+1 into buffer 1
        loadA(Ap, K, tm0, k1, tid, ra);
        compute_ks(&ldsA[0][0], &ldsB[0][0], wm, wn, lane, 0, acc);
        storeA(&ldsA[1][0], tid, ra);
        loadB(Bp, N, tn0, k1, tid, rb);
        compute_ks(&ldsA[0][0], &ldsB[0][0], wm, wn, lane, 1, acc);
        storeB(&ldsB[1][0], tid, rb);
        __syncthreads();

        // half 2: compute buffer 1, stage tile kt+2 (or dummy) into buffer 0
        loadA(Ap, K, tm0, k2, tid, ra);
        compute_ks(&ldsA[1][0], &ldsB[1][0], wm, wn, lane, 0, acc);
        storeA(&ldsA[0][0], tid, ra);
        loadB(Bp, N, tn0, k2, tid, rb);
        compute_ks(&ldsA[1][0], &ldsB[1][0], wm, wn, lane, 1, acc);
        storeB(&ldsB[0][0], tid, rb);
        __syncthreads();
    }

    // --- epilogue: C/D layout -> row = vgpr + (lane>=16 ? 8 : 0), col = lane&15 ---
    const int r0 = tm0 + wm * 64;
    const int c0 = tn0 + wn * 32;
    #pragma unroll
    for (int mf = 0; mf < 4; ++mf) {
        #pragma unroll
        for (int nf = 0; nf < 2; ++nf) {
            v8f a = acc[mf][nf];
            const int col   = c0 + nf * 16 + (lane & 15);
            const int rbase = r0 + mf * 16 + ((lane >> 4) << 3);
            #pragma unroll
            for (int r = 0; r < 8; ++r) {
                const int grow = rbase + r;
                const size_t idx = ((size_t)bz * M + grow) * (size_t)N + col;
                float v = a[r];
                if (mode == 0) {
                    out_bf[idx] = f2bf(gelu_erf(v + bias[col]));
                } else if (mode == 1) {
                    v = gelu_erf(v + rowbias[grow] + bf2f(res_bf[idx]));
                    out_bf[idx] = f2bf(v);
                } else {
                    out_f32[idx] = res_f32[idx] + gelu_erf(v + bias[col]);
                }
            }
        }
    }
}

// ---------- host side ----------
extern "C" void kernel_launch(void* const* d_in, const int* in_sizes, int n_in,
                              void* d_out, int out_size, void* d_ws, size_t ws_size,
                              hipStream_t stream) {
    (void)in_sizes; (void)n_in; (void)out_size; (void)ws_size;
    const int B = 8, S = 2048, D = 768, F = 3072;
    const int BS = B * S;                       // 16384 rows

    const float* x     = (const float*)d_in[0];
    const float* ln1_g = (const float*)d_in[1];
    const float* ln1_b = (const float*)d_in[2];
    const float* W1    = (const float*)d_in[3];
    const float* b1    = (const float*)d_in[4];
    const float* ln2_g = (const float*)d_in[5];
    const float* ln2_b = (const float*)d_in[6];
    const float* Wc    = (const float*)d_in[7];
    const float* bc    = (const float*)d_in[8];
    const float* Sp_i  = (const float*)d_in[9];
    const float* bp_i  = (const float*)d_in[10];
    const float* Sp_ii = (const float*)d_in[11];
    const float* bp_ii = (const float*)d_in[12];
    const float* W2    = (const float*)d_in[13];
    const float* b2    = (const float*)d_in[14];
    float* out = (float*)d_out;

    // workspace layout (all offsets 256B aligned)
    char* w = (char*)d_ws;
    u16* xln_bf  = (u16*)w;                      w += (size_t)BS * D * 2;      // 25.2 MB
    u16* W1_bf   = (u16*)w;                      w += (size_t)D * F * 2;       //  4.7 MB
    u16* h_bf    = (u16*)w;                      w += (size_t)BS * F * 2;      // 100.7 MB
    u16* hn_bf   = (u16*)w;                      w += (size_t)BS * F * 2;      // 100.7 MB
    u16* spi_bf  = (u16*)w;                      w += (size_t)S * S * 2;       //  8.4 MB
    u16* spii_bf = (u16*)w;                      w += (size_t)S * S * 2;       //  8.4 MB
    u16* g_bf    = (u16*)w;                      w += (size_t)BS * F * 2;      // 100.7 MB
    u16* W2_bf   = (u16*)w;                      w += (size_t)F * D * 2;       //  4.7 MB
    int* route   = (int*)w;                      w += 256;

    dim3 blk(256);

    // 1) weight conversions to bf16
    cvt_f32_bf16<<<dim3((D * F) / 256), blk, 0, stream>>>(W1, W1_bf, D * F);
    cvt_f32_bf16<<<dim3((F * D) / 256), blk, 0, stream>>>(W2, W2_bf, F * D);
    cvt_f32_bf16<<<dim3((S * S) / 256), blk, 0, stream>>>(Sp_i,  spi_bf,  S * S);
    cvt_f32_bf16<<<dim3((S * S) / 256), blk, 0, stream>>>(Sp_ii, spii_bf, S * S);

    // 2) LN1: x -> xln (bf16)
    ln_f32<<<dim3(BS), blk, 0, stream>>>(x, ln1_g, ln1_b, xln_bf, D, D / 256);

    // 3) GEMM1: h = gelu(xln @ W1 + b1)   [16384,768]x[768,3072]
    gemm_bf16_wmma<<<dim3(F / 128, BS / 128, 1), blk, 0, stream>>>(
        xln_bf, W1_bf, BS, F, D, /*mode=*/0, b1,
        nullptr, nullptr, h_bf, nullptr,
        nullptr, nullptr, nullptr, nullptr, nullptr);

    // 4) LN2: hn = LN(h)  over F
    ln_bf16<<<dim3(BS), blk, 0, stream>>>(h_bf, ln2_g, ln2_b, hn_bf, F, F / 256);

    // 5) cls routing per batch
    cls_route<<<dim3(B), blk, 0, stream>>>(hn_bf, Wc, bc, route, S, F);

    // 6) spatial GEMM (selected expert only): g = gelu(Sp_sel @ hn + bp_sel + h)
    gemm_bf16_wmma<<<dim3(F / 128, S / 128, B), blk, 0, stream>>>(
        nullptr, hn_bf, S, F, S, /*mode=*/1, nullptr,
        h_bf, nullptr, g_bf, nullptr,
        spi_bf, spii_bf, bp_i, bp_ii, route);

    // 7) GEMM2: out = x + gelu(g @ W2 + b2)  [16384,3072]x[3072,768]
    gemm_bf16_wmma<<<dim3(D / 128, BS / 128, 1), blk, 0, stream>>>(
        g_bf, W2_bf, BS, D, F, /*mode=*/2, b2,
        nullptr, x, nullptr, out,
        nullptr, nullptr, nullptr, nullptr, nullptr);
}
